// QLSTM_57380763075320
// MI455X (gfx1250) — matrix operand order repaired and checked by reading, not compiled, not run on hardware
//
#include <hip/hip_runtime.h>
#include <hip/hip_bf16.h>
#include <math.h>

// ---------------------------------------------------------------------------
// QLSTM for MI455X (gfx1250): all three GEMMs on v_wmma_f32_16x16x32_f16.
// T=256 B=128 F=256 H=512. Recurrence serialized by 256 stream-ordered
// step-kernel launches (graph-capture safe).
// ---------------------------------------------------------------------------

typedef __attribute__((ext_vector_type(16))) _Float16 v16h;
typedef __attribute__((ext_vector_type(8)))  _Float16 v8h;
typedef __attribute__((ext_vector_type(8)))  float    v8f;

#define T_STEPS 256
#define BATCH   128
#define FDIM    256
#define HDIM    512

// Hamilton product tables, indexed [b][a] where
//   K[a*Q + fr][b*Q + hc] = QSIGN[b][a] * w[QCOMP[b][a]][fr][hc]
__device__ __constant__ int   QCOMP[4][4] = {{0,1,2,3},{1,0,3,2},{2,3,0,1},{3,2,1,0}};
__device__ __constant__ float QSIGN[4][4] = {{ 1.f,-1.f,-1.f,-1.f},
                                             { 1.f, 1.f,-1.f, 1.f},
                                             { 1.f, 1.f, 1.f,-1.f},
                                             { 1.f,-1.f, 1.f, 1.f}};

// ---- fragment loaders ------------------------------------------------------
// A (16x32 f16, M x K) from a row-major matrix, contiguous K order (matches packed B).
// lane<16 : row=lane,   K = {0..15}
// lane>=16: row=lane-16, K = {16..31}
__device__ __forceinline__ v16h load_a_rowmajor(const _Float16* __restrict__ base,
                                                int ld, int m0, int k0, int lane) {
    const _Float16* p = base + (size_t)(m0 + (lane & 15)) * ld + k0 + ((lane >> 4) << 4);
    v8h lo = *(const v8h*)(p);
    v8h hi = *(const v8h*)(p + 8);
    v16h r;
#pragma unroll
    for (int i = 0; i < 8; ++i) { r[i] = lo[i]; r[i + 8] = hi[i]; }
    return r;
}

// B (32x16 f16, K x N), pre-packed so each lane's 16 halfs are contiguous.
__device__ __forceinline__ v16h load_b_packed(const _Float16* __restrict__ frag, int lane) {
    return *(const v16h*)(frag + lane * 16);
}

// ---- prep kernels ----------------------------------------------------------
__global__ void qlstm_cvt_x(const float* __restrict__ x, _Float16* __restrict__ x16, int n) {
    int i = blockIdx.x * blockDim.x + threadIdx.x;
    if (i < n) x16[i] = (_Float16)x[i];
}

// Pack Hamilton kernel of w (4 gates, 4 comps, quadIn, 128) into B fragments:
// dst[((g*32 + nt)*ktiles + kt)*512 + lane*16 + i]  (== dst[idx])
// fragment element: K = kt*32 + 16*(lane>>4) + i, N = nt*16 + (lane&15)
__global__ void qlstm_pack_hamilton(const float* __restrict__ w, _Float16* __restrict__ dst,
                                    int total, int ktiles, int quadIn) {
    int idx = blockIdx.x * blockDim.x + threadIdx.x;
    if (idx >= total) return;
    int i    = idx & 15;
    int lane = (idx >> 4) & 31;
    int t2   = idx >> 9;
    int kt   = t2 % ktiles;
    int t3   = t2 / ktiles;
    int nt   = t3 & 31;
    int g    = t3 >> 5;
    int k = kt * 32 + ((lane >> 4) << 4) + i;
    int n = nt * 16 + (lane & 15);
    int a = k / quadIn, fr = k % quadIn;
    int b = n >> 7,     hc = n & 127;
    float v = QSIGN[b][a] * w[((size_t)(g * 4 + QCOMP[b][a]) * quadIn + fr) * 128 + hc];
    dst[idx] = (_Float16)v;
}

// Pack fco_w (F,H) as B with K=H rows, N=F cols: B[k][n] = fco_w[n][k]
__global__ void qlstm_pack_fco(const float* __restrict__ fcow, _Float16* __restrict__ dst, int total) {
    int idx = blockIdx.x * blockDim.x + threadIdx.x;
    if (idx >= total) return;
    int i    = idx & 15;
    int lane = (idx >> 4) & 31;
    int kt   = (idx >> 9) & 15;
    int nt   = idx >> 13;
    int k = kt * 32 + ((lane >> 4) << 4) + i;
    int n = nt * 16 + (lane & 15);
    dst[idx] = (_Float16)fcow[(size_t)n * HDIM + k];
}

__global__ void qlstm_zero(_Float16* __restrict__ h0, float* __restrict__ c, int n) {
    int i = blockIdx.x * blockDim.x + threadIdx.x;
    if (i < n) { h0[i] = (_Float16)0.f; c[i] = 0.f; }
}

// ---- recurrent step --------------------------------------------------------
// grid (B/16, H/64), block 128 (4 waves). Each wave: one 16x16 h tile,
// all 4 gates. A fragments (x_t rows, h_{t-1} rows) reused across gates.
__global__ void __launch_bounds__(128)
qlstm_step(const _Float16* __restrict__ x16,
           const _Float16* __restrict__ KxB,
           const _Float16* __restrict__ KuB,
           const float*    __restrict__ bx,
           _Float16*       __restrict__ hs,
           float*          __restrict__ c,
           int t) {
    const int lane = threadIdx.x & 31;
    const int wave = threadIdx.x >> 5;
    const int m0 = blockIdx.x * 16;
    const int nt = blockIdx.y * 4 + wave;
    const int n0 = nt * 16;

    v8f acc[4] = {};

    // input projection: A = x_t (B x F)
    const _Float16* xA = x16 + (size_t)t * BATCH * FDIM;
#pragma unroll
    for (int kt = 0; kt < FDIM / 32; ++kt) {
        v16h a = load_a_rowmajor(xA, FDIM, m0, kt * 32, lane);
#pragma unroll
        for (int g = 0; g < 4; ++g) {
            v16h b = load_b_packed(KxB + (size_t)((g * 32 + nt) * (FDIM / 32) + kt) * 512, lane);
            acc[g] = __builtin_amdgcn_wmma_f32_16x16x32_f16(false, a, false, b,
                                                            (short)0, acc[g], false, false);
        }
    }
    // recurrent projection: A = h_{t-1} (B x H)
    const _Float16* hA = hs + (size_t)t * BATCH * HDIM;
#pragma unroll
    for (int kt = 0; kt < HDIM / 32; ++kt) {
        v16h a = load_a_rowmajor(hA, HDIM, m0, kt * 32, lane);
#pragma unroll
        for (int g = 0; g < 4; ++g) {
            v16h b = load_b_packed(KuB + (size_t)((g * 32 + nt) * (HDIM / 32) + kt) * 512, lane);
            acc[g] = __builtin_amdgcn_wmma_f32_16x16x32_f16(false, a, false, b,
                                                            (short)0, acc[g], false, false);
        }
    }

    // gate math; C/D layout: vgpr v, lane l -> M = v + 8*(l>=16), N = l%16
    _Float16* hout = hs + (size_t)(t + 1) * BATCH * HDIM;
#pragma unroll
    for (int v = 0; v < 8; ++v) {
        int m = v + ((lane >> 4) << 3);
        int n = n0 + (lane & 15);
        int idx = (m0 + m) * HDIM + n;
        float fpre = acc[0][v] + bx[0 * HDIM + n];
        float ipre = acc[1][v] + bx[1 * HDIM + n];
        float opre = acc[2][v] + bx[2 * HDIM + n];
        float apre = acc[3][v] + bx[3 * HDIM + n];
        float ft = 1.f / (1.f + __expf(-fpre));
        float it = 1.f / (1.f + __expf(-ipre));
        float ot = 1.f / (1.f + __expf(-opre));
        float cn = it * tanhf(apre) + ft * c[idx];
        c[idx] = cn;
        hout[idx] = (_Float16)(ot * tanhf(cn));
    }
}

// ---- output projection -----------------------------------------------------
// out[r, f] = hs[r, :] @ fco_w[f, :] + fco_b[f];  r = t*B + b
__global__ void __launch_bounds__(128)
qlstm_out(const _Float16* __restrict__ hs,
          const _Float16* __restrict__ FcoB,
          const float*    __restrict__ fcob,
          float*          __restrict__ out) {
    const int lane = threadIdx.x & 31;
    const int wave = threadIdx.x >> 5;
    const int m0 = blockIdx.x * 16;
    const int nt = blockIdx.y * 4 + wave;
    const int n0 = nt * 16;

    const _Float16* hA = hs + (size_t)BATCH * HDIM;  // skip h_{-1} zero slab
    v8f acc = {};
#pragma unroll
    for (int kt = 0; kt < HDIM / 32; ++kt) {
        v16h a = load_a_rowmajor(hA, HDIM, m0, kt * 32, lane);
        v16h b = load_b_packed(FcoB + (size_t)(nt * (HDIM / 32) + kt) * 512, lane);
        acc = __builtin_amdgcn_wmma_f32_16x16x32_f16(false, a, false, b,
                                                     (short)0, acc, false, false);
    }
#pragma unroll
    for (int v = 0; v < 8; ++v) {
        int m = v + ((lane >> 4) << 3);
        int n = n0 + (lane & 15);
        out[(size_t)(m0 + m) * FDIM + n] = acc[v] + fcob[n];
    }
}

// ---------------------------------------------------------------------------
extern "C" void kernel_launch(void* const* d_in, const int* in_sizes, int n_in,
                              void* d_out, int out_size, void* d_ws, size_t ws_size,
                              hipStream_t stream) {
    const float* x    = (const float*)d_in[0];  // (T,B,F)
    const float* Wx   = (const float*)d_in[1];  // (4,4,F/4,H/4)
    const float* bx   = (const float*)d_in[2];  // (4,H)
    const float* Uh   = (const float*)d_in[3];  // (4,4,H/4,H/4)
    const float* fcow = (const float*)d_in[4];  // (F,H)
    const float* fcob = (const float*)d_in[5];  // (F,)
    float* out = (float*)d_out;

    char* ws = (char*)d_ws;
    size_t off = 0;
    auto take = [&](size_t bytes) -> char* {
        char* p = ws + off;
        off = (off + bytes + 255) & ~(size_t)255;
        return p;
    };
    _Float16* x16  = (_Float16*)take((size_t)T_STEPS * BATCH * FDIM * 2);      // 16.8 MB
    _Float16* KxB  = (_Float16*)take((size_t)4 * 32 * (FDIM / 32) * 512 * 2);  // 1.0 MB
    _Float16* KuB  = (_Float16*)take((size_t)4 * 32 * (HDIM / 32) * 512 * 2);  // 2.1 MB
    _Float16* FcoB = (_Float16*)take((size_t)(FDIM / 16) * (HDIM / 32) * 512 * 2);
    _Float16* hs   = (_Float16*)take((size_t)(T_STEPS + 1) * BATCH * HDIM * 2); // 33.7 MB
    float*    cbuf = (float*)take((size_t)BATCH * HDIM * 4);

    const int nX = T_STEPS * BATCH * FDIM;
    qlstm_cvt_x<<<(nX + 255) / 256, 256, 0, stream>>>(x, x16, nX);

    const int nKx = 4 * 32 * (FDIM / 32) * 512;
    qlstm_pack_hamilton<<<(nKx + 255) / 256, 256, 0, stream>>>(Wx, KxB, nKx, FDIM / 32, FDIM / 4);

    const int nKu = 4 * 32 * (HDIM / 32) * 512;
    qlstm_pack_hamilton<<<(nKu + 255) / 256, 256, 0, stream>>>(Uh, KuB, nKu, HDIM / 32, HDIM / 4);

    const int nFc = (FDIM / 16) * (HDIM / 32) * 512;
    qlstm_pack_fco<<<(nFc + 255) / 256, 256, 0, stream>>>(fcow, FcoB, nFc);

    const int nZ = BATCH * HDIM;
    qlstm_zero<<<(nZ + 255) / 256, 256, 0, stream>>>(hs, cbuf, nZ);

    // sequential recurrence: stream ordering provides the t -> t+1 dependency
    for (int t = 0; t < T_STEPS; ++t)
        qlstm_step<<<dim3(BATCH / 16, HDIM / 64), 128, 0, stream>>>(
            x16, KxB, KuB, bx, hs, cbuf, t);

    qlstm_out<<<dim3(T_STEPS * BATCH / 16, FDIM / 64), 128, 0, stream>>>(
        hs, FcoB, fcob, out);
}
